// FullTTT_3204045603641
// MI455X (gfx1250) — compile-verified, matrix operand order checked by reading, not run on hardware
//
#include <hip/hip_runtime.h>

// ---------------------------------------------------------------------------
// TTT forward on MI455X (gfx1250, wave32).
//   Kernel 1: prep        — convert embed/ffw1/ffw2 to f16 (one tiny block)
//   Kernel 2: encoder     — per 16-token tile, WMMA f16 GEMMs + bias/relu,
//                            residual + layernorm, writes hs[B][L][H] f32
//   Kernel 3: ttt_scan    — 1 workgroup (64 thr) per batch element; fast
//                            weights live in LDS; 1023 sequential steps;
//                            fused final readout through wout.
// Workspace: hs (128 MB f32) + 40 KB f16 weights.
// ---------------------------------------------------------------------------

typedef __attribute__((ext_vector_type(16))) _Float16 v16h;
typedef __attribute__((ext_vector_type(8)))  float    v8f;

#define Hdim   64
#define INNER  32
#define VOCAB  64
#define BATCH  256
#define SEQLEN 2048
#define NSTEPS 1023   // (SEQLEN-2)/2
#define LRATE  0.05f

// -------------------------------------------------------------- prep kernel
__global__ void prep_f16(const float* __restrict__ embed,
                         const float* __restrict__ ffw1,
                         const float* __restrict__ ffw2,
                         _Float16* __restrict__ embH,
                         _Float16* __restrict__ w1H,
                         _Float16* __restrict__ w2H) {
  for (int j = threadIdx.x; j < VOCAB * Hdim;   j += blockDim.x) embH[j] = (_Float16)embed[j];
  for (int j = threadIdx.x; j < 2 * Hdim * Hdim; j += blockDim.x) w1H[j] = (_Float16)ffw1[j];
  for (int j = threadIdx.x; j < 2 * Hdim * Hdim; j += blockDim.x) w2H[j] = (_Float16)ffw2[j];
}

// ------------------------------------------------------------- encoder
// 128 threads = 4 waves; each wave computes one 16-token tile with WMMA.
// Grid = B*L/64 blocks.
__global__ __launch_bounds__(128) void encoder_wmma(
    const int*      __restrict__ seq,
    const _Float16* __restrict__ embH,
    const float*    __restrict__ embed32,
    const _Float16* __restrict__ w1H,
    const _Float16* __restrict__ w2H,
    const float*    __restrict__ ffb1,
    const float*    __restrict__ ffb2,
    const float*    __restrict__ lng,
    const float*    __restrict__ lnb,
    float*          __restrict__ hs) {
  __shared__ _Float16 sW1[2 * Hdim * Hdim];   // ffw1 [128][64] f16, 16 KB
  __shared__ _Float16 sW2[2 * Hdim * Hdim];   // ffw2 [64][128] f16, 16 KB
  __shared__ float    sU[4 * 16 * Hdim];      // union: mid f16 [wv][16][128] / outb f32 [wv][16][64]
  __shared__ float    sB1[2 * Hdim];
  __shared__ float    sB2[Hdim], sG[Hdim], sBt[Hdim];
  __shared__ int      stid[64];

  const int tid     = threadIdx.x;
  const int lane    = tid & 31;
  const int wv      = tid >> 5;           // wave id 0..3
  const int tokBase = blockIdx.x * 64;    // first token of this block

  { // cooperative stage of f16 weights (dword copies) + biases + token ids
    const uint32_t* s1 = (const uint32_t*)w1H;
    const uint32_t* s2 = (const uint32_t*)w2H;
    uint32_t* d1 = (uint32_t*)sW1;
    uint32_t* d2 = (uint32_t*)sW2;
    for (int j = tid; j < Hdim * Hdim; j += 128) { d1[j] = s1[j]; d2[j] = s2[j]; }
    if (tid < 2 * Hdim) sB1[tid] = ffb1[tid];
    if (tid < Hdim) { sB2[tid] = ffb2[tid]; sG[tid] = lng[tid]; sBt[tid] = lnb[tid]; }
    if (tid < 64) stid[tid] = seq[tokBase + tid];
  }
  __syncthreads();

  const int mN = lane & 15;            // row (A) / col (B,D) index within tile
  const int hi = (lane >> 4) & 1;      // lane-half selector
  _Float16* mid  = (_Float16*)sU + wv * (16 * 128);
  float*    outb = sU + wv * (16 * Hdim);

  // ---- GEMM1: e[16x64] @ ffw1^T -> [16x128] ; bias + relu -> mid (f16)
  v16h afrag[2];
  {
    int tok = stid[wv * 16 + mN];
#pragma unroll
    for (int kc = 0; kc < 2; ++kc)
#pragma unroll
      for (int j = 0; j < 16; ++j) {
        int kk = (j & 7) + ((j >> 3) << 4) + (hi << 3);   // 16-bit A/B frag K index
        afrag[kc][j] = embH[tok * Hdim + kc * 32 + kk];
      }
  }
#pragma unroll
  for (int nt = 0; nt < 8; ++nt) {
    v8f acc = {};
#pragma unroll
    for (int kc = 0; kc < 2; ++kc) {
      v16h bfrag;
#pragma unroll
      for (int j = 0; j < 16; ++j) {
        int kk = (j & 7) + ((j >> 3) << 4) + (hi << 3);
        bfrag[j] = sW1[(nt * 16 + mN) * Hdim + kc * 32 + kk];   // B[k][n] = ffw1[n][k]
      }
      acc = __builtin_amdgcn_wmma_f32_16x16x32_f16(false, afrag[kc], false, bfrag,
                                                   (short)0, acc, false, false);
    }
#pragma unroll
    for (int r = 0; r < 8; ++r) {
      int m = r + hi * 8;
      int n = nt * 16 + mN;
      float x = acc[r] + sB1[n];
      mid[m * 128 + n] = (_Float16)fmaxf(x, 0.f);
    }
  }
  __syncthreads();

  // ---- GEMM2 A-fragments from mid (then mid's LDS is reused as outb)
  v16h a2[4];
#pragma unroll
  for (int kc = 0; kc < 4; ++kc)
#pragma unroll
    for (int j = 0; j < 16; ++j) {
      int kk = (j & 7) + ((j >> 3) << 4) + (hi << 3);
      a2[kc][j] = mid[mN * 128 + kc * 32 + kk];
    }
  __syncthreads();   // all mid reads done; outb may alias

  // ---- GEMM2: mid[16x128] @ ffw2^T -> [16x64] ; + bias -> outb (f32)
#pragma unroll
  for (int nt = 0; nt < 4; ++nt) {
    v8f acc = {};
#pragma unroll
    for (int kc = 0; kc < 4; ++kc) {
      v16h bfrag;
#pragma unroll
      for (int j = 0; j < 16; ++j) {
        int kk = (j & 7) + ((j >> 3) << 4) + (hi << 3);
        bfrag[j] = sW2[(nt * 16 + mN) * (2 * Hdim) + kc * 32 + kk]; // B[k][n] = ffw2[n][k]
      }
      acc = __builtin_amdgcn_wmma_f32_16x16x32_f16(false, a2[kc], false, bfrag,
                                                   (short)0, acc, false, false);
    }
#pragma unroll
    for (int r = 0; r < 8; ++r) {
      int m = r + hi * 8;
      int n = nt * 16 + mN;
      outb[m * Hdim + n] = acc[r] + sB2[n];
    }
  }
  __syncthreads();

  // ---- residual + layernorm: lanes 0..15 each own one token row
  if (lane < 16) {
    int  m    = lane;
    int  tok  = stid[wv * 16 + m];
    long gtok = (long)(tokBase + wv * 16 + m);
    const float* erow = embed32 + tok * Hdim;
    float s1 = 0.f, s2 = 0.f;
#pragma unroll 8
    for (int h = 0; h < Hdim; ++h) {
      float xv = outb[m * Hdim + h] + erow[h];
      s1 += xv; s2 += xv * xv;
    }
    float mu  = s1 * (1.f / Hdim);
    float var = s2 * (1.f / Hdim) - mu * mu;
    float inv = rsqrtf(var + 1e-5f);
#pragma unroll 8
    for (int h = 0; h < Hdim; ++h) {
      float xv = outb[m * Hdim + h] + erow[h];
      hs[gtok * Hdim + h] = (xv - mu) * inv * sG[h] + sBt[h];
    }
  }
}

// ------------------------------------------------------------- TTT scan
// One workgroup (64 threads, 2 waves) per batch element; state in LDS.
__global__ __launch_bounds__(64) void ttt_scan(
    const float* __restrict__ hs,
    const float* __restrict__ w1_0, const float* __restrict__ b1_0,
    const float* __restrict__ w2_0, const float* __restrict__ b2_0,
    const float* __restrict__ wout, const float* __restrict__ bout,
    float* __restrict__ out) {
  __shared__ float sw1[INNER][Hdim + 1];    // 32 x 65 (pad: conflict-free rows)
  __shared__ float sw2[Hdim][INNER + 1];    // 64 x 33 (pad: conflict-free cols)
  __shared__ float sb1[INNER], sb2[Hdim];
  __shared__ float sk[Hdim], sv[Hdim];
  __shared__ float spre[INNER], sh[INNER], sdh[INNER];
  __shared__ float sd[Hdim];

  const int t = threadIdx.x;            // 0..63
  const int b = blockIdx.x;             // batch element
  const float* hsb = hs + (size_t)b * SEQLEN * Hdim;

  for (int idx = t; idx < INNER * Hdim; idx += 64) sw1[idx >> 6][idx & 63] = w1_0[idx];
  for (int idx = t; idx < Hdim * INNER; idx += 64) sw2[idx >> 5][idx & 31] = w2_0[idx];
  if (t < INNER) sb1[t] = b1_0[t];
  sb2[t] = b2_0[t];
  __syncthreads();

  for (int s = 0; s < NSTEPS; ++s) {
    // k = hs row 2s, v = hs row 2s+1 (contiguous 512B, coalesced, L2-hot)
    sk[t] = hsb[(size_t)s * 128 + t];
    sv[t] = hsb[(size_t)s * 128 + 64 + t];
    __builtin_prefetch(&hsb[(size_t)(s + 1) * 128 + t], 0, 1);  // next step's k/v
    __syncthreads();

    if (t < INNER) {                    // pre = w1@k + b1 ; h = relu(pre)
      float a = sb1[t];
#pragma unroll 8
      for (int j = 0; j < Hdim; ++j) a = fmaf(sw1[t][j], sk[j], a);
      spre[t] = a;
      sh[t]   = fmaxf(a, 0.f);
    }
    __syncthreads();

    {                                   // d = 2/H * (w2@h + b2 - v)
      float a = sb2[t];
#pragma unroll 8
      for (int j = 0; j < INNER; ++j) a = fmaf(sw2[t][j], sh[j], a);
      sd[t] = (2.f / Hdim) * (a - sv[t]);
    }
    __syncthreads();

    if (t < INNER) {                    // dh = (w2^T @ d) * (pre>0)
      float a = 0.f;
#pragma unroll 8
      for (int j = 0; j < Hdim; ++j) a = fmaf(sw2[j][t], sd[j], a);
      sdh[t] = (spre[t] > 0.f) ? a : 0.f;
    }
    __syncthreads();

    // rank-1 SGD updates, spread over all 64 lanes
#pragma unroll 4
    for (int idx = t; idx < INNER * Hdim; idx += 64) {
      int r = idx >> 6, c = idx & 63;
      sw1[r][c] = fmaf(-LRATE * sdh[r], sk[c], sw1[r][c]);
    }
#pragma unroll 4
    for (int idx = t; idx < Hdim * INNER; idx += 64) {
      int r = idx >> 5, c = idx & 31;
      sw2[r][c] = fmaf(-LRATE * sd[r], sh[c], sw2[r][c]);
    }
    if (t < INNER) sb1[t] -= LRATE * sdh[t];
    sb2[t] -= LRATE * sd[t];
    __syncthreads();
  }

  // ---- final query with updated weights: ctx = w2f@relu(w1f@last+b1f)+b2f
  sk[t] = hsb[(size_t)(SEQLEN - 1) * Hdim + t];
  __syncthreads();
  if (t < INNER) {
    float a = sb1[t];
#pragma unroll 8
    for (int j = 0; j < Hdim; ++j) a = fmaf(sw1[t][j], sk[j], a);
    sh[t] = fmaxf(a, 0.f);
  }
  __syncthreads();
  {
    float a = sb2[t];
#pragma unroll 8
    for (int j = 0; j < INNER; ++j) a = fmaf(sw2[t][j], sh[j], a);
    sd[t] = a;                          // ctx vector
  }
  __syncthreads();
  {                                     // out[b][v] = ctx . wout[v,:] + bout[v]
    float a = bout[t];
#pragma unroll 8
    for (int h = 0; h < Hdim; ++h) a = fmaf(sd[h], wout[t * Hdim + h], a);
    out[b * VOCAB + t] = a;
  }
}

// ------------------------------------------------------------- launch
extern "C" void kernel_launch(void* const* d_in, const int* in_sizes, int n_in,
                              void* d_out, int out_size, void* d_ws, size_t ws_size,
                              hipStream_t stream) {
  const int*   seq   = (const int*)  d_in[0];
  const float* embed = (const float*)d_in[1];
  const float* ffw1  = (const float*)d_in[2];
  const float* ffb1  = (const float*)d_in[3];
  const float* ffw2  = (const float*)d_in[4];
  const float* ffb2  = (const float*)d_in[5];
  const float* ln_g  = (const float*)d_in[6];
  const float* ln_b  = (const float*)d_in[7];
  const float* w1    = (const float*)d_in[8];
  const float* b1    = (const float*)d_in[9];
  const float* w2    = (const float*)d_in[10];
  const float* b2    = (const float*)d_in[11];
  const float* wout  = (const float*)d_in[12];
  const float* bout  = (const float*)d_in[13];
  float* out = (float*)d_out;

  // workspace: hs f32 [B][L][H] then f16 weight copies
  float*    hsbuf = (float*)d_ws;
  size_t    hsBytes = (size_t)BATCH * SEQLEN * Hdim * sizeof(float);
  _Float16* embH = (_Float16*)((char*)d_ws + hsBytes);
  _Float16* w1H  = embH + VOCAB * Hdim;
  _Float16* w2H  = w1H + 2 * Hdim * Hdim;

  prep_f16<<<1, 256, 0, stream>>>(embed, ffw1, ffw2, embH, w1H, w2H);

  int encBlocks = (BATCH * SEQLEN) / 64;   // 8192, 4 tiles of 16 tokens per block
  encoder_wmma<<<encBlocks, 128, 0, stream>>>(seq, embH, embed, w1H, w2H,
                                              ffb1, ffb2, ln_g, ln_b, hsbuf);

  ttt_scan<<<BATCH, 64, 0, stream>>>(hsbuf, w1, b1, w2, b2, wout, bout, out);
}
